// DTIHGAT_34196529611171
// MI455X (gfx1250) — matrix operand order repaired and checked by reading, not compiled, not run on hardware
//
#include <hip/hip_runtime.h>
#include <math.h>

// ---------------- problem constants (from reference) ----------------
#define NPn 100000
#define NLn 50000
#define EPn 1600000
#define ELn 800000
#define Hh  2
#define Bb  64
#define IMG_H 140
#define IMG_W 31

typedef __attribute__((ext_vector_type(16))) _Float16 v16h;
typedef __attribute__((ext_vector_type(8)))  _Float16 v8h;
typedef __attribute__((ext_vector_type(8)))  float    v8f;

static inline int cdiv(int a, int b) { return (a + b - 1) / b; }

// ---------------- workspace layout (float offsets) ----------------
static const size_t OFF_FBA   = 0;               // ping node features (<= 50000*74)
static const size_t OFF_FBB   = 4000000;         // pong node features
static const size_t OFF_HF    = 8000000;         // h = x@W   (<= 50000*140)
static const size_t OFF_OA    = 15200000;        // scatter accumulator
static const size_t OFF_EL    = 22400000;        // el[n,h]
static const size_t OFF_ER    = 22600704;        // er[n,h]
static const size_t OFF_MM    = 22801408;        // seg-max[n,h]  (reused as gmax[B])
static const size_t OFF_SS    = 23002112;        // seg-sum[n,h]  (reused as gsum[B])
static const size_t OFF_EV    = 23202816;        // edge logits [E,h]
static const size_t OFF_AV    = 26402816;        // edge exp    [E,h]
static const size_t OFF_POOL  = 29602816;        // pooled: lig[64*31] then prot[64*31]
static const size_t OFF_IMG   = 29606912;        // 140*31
static const size_t OFF_T1    = 29611264;        // 64*140*31
static const size_t OFF_T2    = 29889024;
static const size_t OFF_T3    = 30166784;
static const size_t OFF_F64   = 30444544;
static const size_t OFF_F512  = 30444608;
static const size_t OFF_V512  = 30445120;
static const size_t OFF_O512  = 30445632;
static const size_t OFF_T4340 = 30446144;

// ---------------- device helpers ----------------
static __device__ __forceinline__ void atomicMaxF(float* addr, float v) {
  if (v >= 0.0f) atomicMax((int*)addr, __float_as_int(v));
  else           atomicMin((unsigned int*)addr, __float_as_uint(v));
}

__global__ void k_fill(float* __restrict__ p, float v, int n) {
  int i = blockIdx.x * blockDim.x + threadIdx.x;
  if (i < n) p[i] = v;
}

// ---------------------------------------------------------------------------
// C[M,N] = A[M,K] @ B[K,N], fp32 in/out, f16 WMMA compute, f32 accumulate.
// Compile-time K,N. M must be a multiple of 16 (true: 100000, 50000).
// Block = 128 threads (4 waves); block covers 64 rows and ALL column tiles.
// A tile (64 x KPAD) and B (transposed, NPAD x KPAD) staged in LDS as f16,
// zero-padded, so fragment reads are unconditional 16B ds loads.
// ---------------------------------------------------------------------------
template<int K, int N>
__global__ void __launch_bounds__(128)
k_gemm_wmma_t(const float* __restrict__ A, const float* __restrict__ Bw,
              float* __restrict__ C, int M) {
  constexpr int KT   = (K + 31) / 32;
  constexpr int KPAD = KT * 32;
  constexpr int NT   = (N + 15) / 16;
  constexpr int NPAD = NT * 16;
  __shared__ __align__(16) _Float16 As[64 * KPAD];
  __shared__ __align__(16) _Float16 Bt[NPAD * KPAD];

  const int tid  = threadIdx.x;
  const int lane = tid & 31;
  const int wave = tid >> 5;
  const int blockRow0 = blockIdx.x * 64;

  // stage A tile: k-fastest -> coalesced global reads
  for (int idx = tid; idx < 64 * KPAD; idx += 128) {
    int r = idx / KPAD, k = idx % KPAD;
    int m = blockRow0 + r;
    float v = (m < M && k < K) ? A[(size_t)m * K + k] : 0.f;
    As[idx] = (_Float16)v;
  }
  // stage B transposed: read n-fastest (coalesced), write [n][k]
  for (int idx = tid; idx < NPAD * KPAD; idx += 128) {
    int k = idx / NPAD, n = idx % NPAD;
    float v = (k < K && n < N) ? Bw[(size_t)k * N + n] : 0.f;
    Bt[n * KPAD + k] = (_Float16)v;
  }
  __syncthreads();

  const int row0 = blockRow0 + wave * 16;
  if (row0 >= M) return;               // wave-uniform: EXEC stays full for WMMA

  const int ml = lane & 15;
  const int hf = lane >> 4;

  v8f acc[NT];
  #pragma unroll
  for (int t = 0; t < NT; ++t) acc[t] = {};

  #pragma unroll
  for (int kt = 0; kt < KT; ++kt) {
    // A fragment: af[i] = A[m][kt*32 + hf*8 + i] (i<8), af[8+i] = A[m][kt*32+16+hf*8+i]
    const _Float16* ap = &As[(wave * 16 + ml) * KPAD + kt * 32 + hf * 8];
    v8h a0 = *(const v8h*)ap;
    v8h a1 = *(const v8h*)(ap + 16);
    v16h af = __builtin_shufflevector(a0, a1, 0,1,2,3,4,5,6,7,8,9,10,11,12,13,14,15);
    #pragma unroll
    for (int t = 0; t < NT; ++t) {
      // B fragment: bf[i] = B[kt*32 + hf*8 + i][n] (i<8), bf[8+i] = B[kt*32+16+hf*8+i][n]
      const _Float16* bp = &Bt[(t * 16 + ml) * KPAD + kt * 32 + hf * 8];
      v8h b0 = *(const v8h*)bp;
      v8h b1 = *(const v8h*)(bp + 16);
      v16h bf = __builtin_shufflevector(b0, b1, 0,1,2,3,4,5,6,7,8,9,10,11,12,13,14,15);
      acc[t] = __builtin_amdgcn_wmma_f32_16x16x32_f16(false, af, false, bf,
                                                      (short)0, acc[t], false, false);
    }
  }

  #pragma unroll
  for (int t = 0; t < NT; ++t) {
    int n = t * 16 + ml;
    if (n < N) {
      #pragma unroll
      for (int v = 0; v < 8; ++v) {        // C layout: m = v + 8*half
        int m = row0 + v + hf * 8;
        C[(size_t)m * N + n] = acc[t][v];
      }
    }
  }
}

// el[n,h] = <h[n,h,:], al[h,:]> ; er likewise
template<int D>
__global__ void k_attn_coef(const float* __restrict__ hfm, const float* __restrict__ al,
                            const float* __restrict__ ar, float* __restrict__ el,
                            float* __restrict__ er, int N) {
  int i = blockIdx.x * blockDim.x + threadIdx.x;
  if (i >= N * Hh) return;
  int n = i >> 1, h = i & 1;
  const float* hp = hfm + (size_t)n * Hh * D + (size_t)h * D;
  const float* ap = al + h * D;
  const float* bp = ar + h * D;
  float sl = 0.f, sr = 0.f;
  #pragma unroll
  for (int d = 0; d < D; ++d) { float v = hp[d]; sl += v * ap[d]; sr += v * bp[d]; }
  el[i] = sl; er[i] = sr;
}

__global__ void k_edge_max(const float* __restrict__ el, const float* __restrict__ er,
                           const int* __restrict__ src, const int* __restrict__ dst,
                           float* __restrict__ ev, float* __restrict__ mm, int E) {
  int i = blockIdx.x * blockDim.x + threadIdx.x;
  if (i >= E * Hh) return;
  int e = i >> 1, h = i & 1;
  int s = src[e], d = dst[e];
  float x = el[s * Hh + h] + er[d * Hh + h];
  x = x > 0.f ? x : 0.2f * x;                 // leaky_relu(0.2)
  ev[i] = x;
  atomicMaxF(&mm[d * Hh + h], x);
}

__global__ void k_fix_nonfinite(float* __restrict__ p, int n) {
  int i = blockIdx.x * blockDim.x + threadIdx.x;
  if (i >= n) return;
  if (p[i] < -3.0e38f) p[i] = 0.f;            // only untouched (-inf) slots
}

__global__ void k_edge_exp(const float* __restrict__ ev, const float* __restrict__ mm,
                           const int* __restrict__ dst, float* __restrict__ av,
                           float* __restrict__ ss, int E) {
  int i = blockIdx.x * blockDim.x + threadIdx.x;
  if (i >= E * Hh) return;
  int e = i >> 1, h = i & 1;
  int d = dst[e];
  float a = expf(ev[i] - mm[d * Hh + h]);
  av[i] = a;
  atomicAdd(&ss[d * Hh + h], a);
}

template<int D>
__global__ void k_edge_scatter(const float* __restrict__ av, const float* __restrict__ ss,
                               const float* __restrict__ hfm,
                               const int* __restrict__ src, const int* __restrict__ dst,
                               float* __restrict__ oa, int E) {
  int i = blockIdx.x * blockDim.x + threadIdx.x;
  if (i >= E * Hh) return;
  int e = i >> 1, h = i & 1;
  int s = src[e], d = dst[e];
  float alpha = av[i] / (ss[d * Hh + h] + 1e-9f);
  const float* hp = hfm + (size_t)s * Hh * D + (size_t)h * D;
  float*       op = oa  + (size_t)d * Hh * D + (size_t)h * D;
  __builtin_prefetch(hp, 0, 0);               // global_prefetch_b8
  #pragma unroll
  for (int k = 0; k < D; ++k) atomicAdd(&op[k], alpha * hp[k]);
}

// out[n,d] = relu( mean_h elu(oa[n,h,d]) )
template<int D>
__global__ void k_node_post(const float* __restrict__ oa, float* __restrict__ out, int N) {
  int i = blockIdx.x * blockDim.x + threadIdx.x;
  if (i >= N * D) return;
  int n = i / D, d = i % D;
  float a = oa[(size_t)n * Hh * D + d];
  float b = oa[(size_t)n * Hh * D + D + d];
  a = a > 0.f ? a : expf(a) - 1.f;
  b = b > 0.f ? b : expf(b) - 1.f;
  float v = 0.5f * (a + b);
  out[i] = fmaxf(v, 0.f);
}

// ---- global attention pooling (gate dim fixed = 31) ----
__global__ void k_gate(const float* __restrict__ x, const float* __restrict__ Wg,
                       const float* __restrict__ bg, const int* __restrict__ gid,
                       float* __restrict__ gate, float* __restrict__ gmax, int N) {
  int n = blockIdx.x * blockDim.x + threadIdx.x;
  if (n >= N) return;
  float s = bg[0];
  #pragma unroll
  for (int d = 0; d < 31; ++d) s += x[(size_t)n * 31 + d] * Wg[d];
  gate[n] = s;
  atomicMaxF(&gmax[gid[n]], s);
}

__global__ void k_gate_exp(const float* __restrict__ gate, const float* __restrict__ gmax,
                           const int* __restrict__ gid, float* __restrict__ ga,
                           float* __restrict__ gsum, int N) {
  int n = blockIdx.x * blockDim.x + threadIdx.x;
  if (n >= N) return;
  float a = expf(gate[n] - gmax[gid[n]]);
  ga[n] = a;
  atomicAdd(&gsum[gid[n]], a);
}

__global__ void k_gate_scatter(const float* __restrict__ ga, const float* __restrict__ gsum,
                               const float* __restrict__ x, const int* __restrict__ gid,
                               float* __restrict__ pooled, int N) {
  int i = blockIdx.x * blockDim.x + threadIdx.x;
  if (i >= N * 31) return;
  int n = i / 31, d = i % 31;
  int g = gid[n];
  float alpha = ga[n] / (gsum[g] + 1e-9f);
  atomicAdd(&pooled[g * 31 + d], alpha * x[(size_t)n * 31 + d]);
}

// ---- image assembly / CNN / dense tail ----
__global__ void k_build_img(const float* __restrict__ lig, const float* __restrict__ prot,
                            float* __restrict__ img) {
  int i = blockIdx.x * blockDim.x + threadIdx.x;
  if (i >= IMG_H * IMG_W) return;
  int r = i / IMG_W, c = i % IMG_W;
  float v = 0.f;
  if (r < Bb)          v = lig[r * IMG_W + c];
  else if (r < 2 * Bb) v = prot[(r - Bb) * IMG_W + c];
  img[i] = v;
}

__global__ void k_conv3x3(const float* __restrict__ in, const float* __restrict__ w,
                          const float* __restrict__ bias, const float* __restrict__ res,
                          float* __restrict__ out, int Cin, int Cout, int doRelu) {
  int i = blockIdx.x * blockDim.x + threadIdx.x;
  int total = Cout * IMG_H * IMG_W;
  if (i >= total) return;
  int xw = i % IMG_W;
  int yh = (i / IMG_W) % IMG_H;
  int co = i / (IMG_W * IMG_H);
  float acc = bias[co];
  for (int ci = 0; ci < Cin; ++ci) {
    const float* wp = w + ((size_t)co * Cin + ci) * 9;
    const float* ip = in + (size_t)ci * IMG_H * IMG_W;
    #pragma unroll
    for (int ky = 0; ky < 3; ++ky) {
      int yy = yh + ky - 1;
      if (yy < 0 || yy >= IMG_H) continue;
      #pragma unroll
      for (int kx = 0; kx < 3; ++kx) {
        int xx = xw + kx - 1;
        if (xx < 0 || xx >= IMG_W) continue;
        acc += wp[ky * 3 + kx] * ip[yy * IMG_W + xx];
      }
    }
  }
  if (res) acc += res[i];
  if (doRelu) acc = fmaxf(acc, 0.f);
  out[i] = acc;
}

__global__ void k_chanmean(const float* __restrict__ in, float* __restrict__ out) {
  int c = threadIdx.x;
  if (c >= 64) return;
  float s = 0.f;
  const float* p = in + (size_t)c * IMG_H * IMG_W;
  for (int i = 0; i < IMG_H * IMG_W; ++i) s += p[i];
  out[c] = s * (1.0f / (IMG_H * IMG_W));
}

// y[n] = act(b[n] + sum_k x[k]*W[k,n]); act 0=none 1=relu 2=sigmoid
__global__ void k_gemv(const float* __restrict__ x, const float* __restrict__ W,
                       const float* __restrict__ b, float* __restrict__ y,
                       int K, int N, int act) {
  int n = blockIdx.x * blockDim.x + threadIdx.x;
  if (n >= N) return;
  float s = b ? b[n] : 0.f;
  for (int k = 0; k < K; ++k) s += x[k] * W[(size_t)k * N + n];
  if (act == 1)      s = fmaxf(s, 0.f);
  else if (act == 2) s = 1.f / (1.f + expf(-s));
  y[n] = s;
}

// ---------------- host orchestration ----------------
struct GatWS { float *hf, *oa, *el, *er, *mm, *ss, *ev, *av; };

template<int Din, int Dout>
static void run_gat(const float* x, int N,
                    const float* W, const float* al, const float* ar,
                    const int* src, const int* dst, int E,
                    float* out, const GatWS& w, hipStream_t s) {
  constexpr int HD = Hh * Dout;
  k_gemm_wmma_t<Din, HD><<<cdiv(N, 64), 128, 0, s>>>(x, W, w.hf, N);
  k_attn_coef<Dout><<<cdiv(N * Hh, 256), 256, 0, s>>>(w.hf, al, ar, w.el, w.er, N);
  k_fill<<<cdiv(N * Hh, 256), 256, 0, s>>>(w.mm, -INFINITY, N * Hh);
  k_fill<<<cdiv(N * Hh, 256), 256, 0, s>>>(w.ss, 0.f, N * Hh);
  k_fill<<<cdiv(N * HD, 256), 256, 0, s>>>(w.oa, 0.f, N * HD);
  k_edge_max<<<cdiv(E * Hh, 256), 256, 0, s>>>(w.el, w.er, src, dst, w.ev, w.mm, E);
  k_fix_nonfinite<<<cdiv(N * Hh, 256), 256, 0, s>>>(w.mm, N * Hh);
  k_edge_exp<<<cdiv(E * Hh, 256), 256, 0, s>>>(w.ev, w.mm, dst, w.av, w.ss, E);
  k_edge_scatter<Dout><<<cdiv(E * Hh, 256), 256, 0, s>>>(w.av, w.ss, w.hf, src, dst, w.oa, E);
  k_node_post<Dout><<<cdiv(N * Dout, 256), 256, 0, s>>>(w.oa, out, N);
}

static void run_gap(const float* x, int N, const int* gid,
                    const float* Wg, const float* bg, float* pooled,
                    float* gate, float* ga, float* gmax, float* gsum, hipStream_t s) {
  k_fill<<<1, 64, 0, s>>>(gmax, -INFINITY, Bb);
  k_fill<<<1, 64, 0, s>>>(gsum, 0.f, Bb);
  k_fill<<<cdiv(Bb * 31, 256), 256, 0, s>>>(pooled, 0.f, Bb * 31);
  k_gate<<<cdiv(N, 256), 256, 0, s>>>(x, Wg, bg, gid, gate, gmax, N);
  k_fix_nonfinite<<<1, 64, 0, s>>>(gmax, Bb);
  k_gate_exp<<<cdiv(N, 256), 256, 0, s>>>(gate, gmax, gid, ga, gsum, N);
  k_gate_scatter<<<cdiv(N * 31, 256), 256, 0, s>>>(ga, gsum, x, gid, pooled, N);
}

extern "C" void kernel_launch(void* const* d_in, const int* in_sizes, int n_in,
                              void* d_out, int out_size, void* d_ws, size_t ws_size,
                              hipStream_t stream) {
  (void)in_sizes; (void)n_in; (void)out_size; (void)ws_size;
  // ---- inputs (pytree-leaf order of setup_inputs) ----
  const float* x_p = (const float*)d_in[0];
  const float* x_l = (const float*)d_in[1];
  // 2..16: prot (W,al,ar)*5 ; 17..31: lig (W,al,ar)*5
  const float* pool_p_W = (const float*)d_in[32];
  const float* pool_p_b = (const float*)d_in[33];
  const float* pool_l_W = (const float*)d_in[34];
  const float* pool_l_b = (const float*)d_in[35];
  const float* c1w = (const float*)d_in[36]; const float* c1b = (const float*)d_in[37];
  const float* r1w = (const float*)d_in[38]; const float* r1b = (const float*)d_in[39];
  const float* r2w = (const float*)d_in[40]; const float* r2b = (const float*)d_in[41];
  const float* fw  = (const float*)d_in[42]; const float* fb  = (const float*)d_in[43];
  const float* Wv = (const float*)d_in[48];  const float* bv = (const float*)d_in[49];
  const float* Wo = (const float*)d_in[50];  const float* bo = (const float*)d_in[51];
  const float* fc_in_w  = (const float*)d_in[52];
  const float* fc_in_b  = (const float*)d_in[53];
  const float* fc_out_w = (const float*)d_in[54];
  const float* fc_out_b = (const float*)d_in[55];
  const int* src_p = (const int*)d_in[56];
  const int* dst_p = (const int*)d_in[57];
  const int* gid_p = (const int*)d_in[58];
  const int* src_l = (const int*)d_in[59];
  const int* dst_l = (const int*)d_in[60];
  const int* gid_l = (const int*)d_in[61];

  float* ws = (float*)d_ws;
  float* fbA = ws + OFF_FBA;
  float* fbB = ws + OFF_FBB;
  GatWS g;
  g.hf = ws + OFF_HF;  g.oa = ws + OFF_OA;
  g.el = ws + OFF_EL;  g.er = ws + OFF_ER;
  g.mm = ws + OFF_MM;  g.ss = ws + OFF_SS;
  g.ev = ws + OFF_EV;  g.av = ws + OFF_AV;
  float* pooled_lig  = ws + OFF_POOL;
  float* pooled_prot = ws + OFF_POOL + Bb * 31;
  float* img  = ws + OFF_IMG;
  float* t1   = ws + OFF_T1;
  float* t2   = ws + OFF_T2;
  float* t3   = ws + OFF_T3;
  float* f64  = ws + OFF_F64;
  float* f512 = ws + OFF_F512;
  float* v512 = ws + OFF_V512;
  float* o512 = ws + OFF_O512;
  float* t4340 = ws + OFF_T4340;

  // ---- protein GAT stack (31 -> 31, 5 layers) ----
  {
    const float* cur = x_p;
    for (int l = 0; l < 5; ++l) {
      const float* W  = (const float*)d_in[2 + 3 * l];
      const float* al = (const float*)d_in[3 + 3 * l];
      const float* ar = (const float*)d_in[4 + 3 * l];
      float* out = (l & 1) ? fbB : fbA;
      run_gat<31, 31>(cur, NPn, W, al, ar, src_p, dst_p, EPn, out, g, stream);
      cur = out;
    }
    run_gap(cur, NPn, gid_p, pool_p_W, pool_p_b, pooled_prot,
            g.el, g.er, g.mm, g.ss, stream);
  }

  // ---- ligand GAT stack (74->70->65->60->55->31) ----
  {
    const float* W0  = (const float*)d_in[17]; const float* al0 = (const float*)d_in[18];
    const float* ar0 = (const float*)d_in[19];
    const float* W1  = (const float*)d_in[20]; const float* al1 = (const float*)d_in[21];
    const float* ar1 = (const float*)d_in[22];
    const float* W2  = (const float*)d_in[23]; const float* al2 = (const float*)d_in[24];
    const float* ar2 = (const float*)d_in[25];
    const float* W3  = (const float*)d_in[26]; const float* al3 = (const float*)d_in[27];
    const float* ar3 = (const float*)d_in[28];
    const float* W4  = (const float*)d_in[29]; const float* al4 = (const float*)d_in[30];
    const float* ar4 = (const float*)d_in[31];
    run_gat<74, 70>(x_l, NLn, W0, al0, ar0, src_l, dst_l, ELn, fbA, g, stream);
    run_gat<70, 65>(fbA, NLn, W1, al1, ar1, src_l, dst_l, ELn, fbB, g, stream);
    run_gat<65, 60>(fbB, NLn, W2, al2, ar2, src_l, dst_l, ELn, fbA, g, stream);
    run_gat<60, 55>(fbA, NLn, W3, al3, ar3, src_l, dst_l, ELn, fbB, g, stream);
    run_gat<55, 31>(fbB, NLn, W4, al4, ar4, src_l, dst_l, ELn, fbA, g, stream);
    run_gap(fbA, NLn, gid_l, pool_l_W, pool_l_b, pooled_lig,
            g.el, g.er, g.mm, g.ss, stream);
  }

  // ---- assemble padded image, residual CNN ----
  k_build_img<<<cdiv(IMG_H * IMG_W, 256), 256, 0, stream>>>(pooled_lig, pooled_prot, img);
  const int convN = 64 * IMG_H * IMG_W;
  k_conv3x3<<<cdiv(convN, 256), 256, 0, stream>>>(img, c1w, c1b, nullptr, t1, 1, 64, 1);
  k_conv3x3<<<cdiv(convN, 256), 256, 0, stream>>>(t1, r1w, r1b, nullptr, t2, 64, 64, 1);
  k_conv3x3<<<cdiv(convN, 256), 256, 0, stream>>>(t2, r2w, r2b, t1, t3, 64, 64, 1);
  k_chanmean<<<1, 64, 0, stream>>>(t3, f64);
  k_gemv<<<cdiv(512, 128), 128, 0, stream>>>(f64, fw, fb, f512, 64, 512, 0);

  // ---- MHA with seq len 1: softmax over 1 key == 1  =>  out = (x@Wv+bv)@Wo+bo ----
  k_gemv<<<cdiv(512, 128), 128, 0, stream>>>(f512, Wv, bv, v512, 512, 512, 0);
  k_gemv<<<cdiv(512, 128), 128, 0, stream>>>(v512, Wo, bo, o512, 512, 512, 0);

  // ---- dense tail ----
  k_gemv<<<cdiv(4340, 128), 128, 0, stream>>>(o512, fc_in_w, fc_in_b, t4340, 512, 4340, 1);
  k_gemv<<<1, 32, 0, stream>>>(t4340, fc_out_w, fc_out_b, (float*)d_out, 4340, 1, 2);
}
